// HeteroGraphTransformerLayer2_41747082117530
// MI455X (gfx1250) — matrix operand order repaired and checked by reading, not compiled, not run on hardware
//
#include <hip/hip_runtime.h>
#include <hip/hip_bf16.h>

#define NU 100000
#define NI 50000
#define FDIM 256
#define HH 8
#define DD 64
#define NOUT 512   // H*D
#define E1 300000
#define E2 300000
#define SLOPE 0.2f

typedef __attribute__((ext_vector_type(16))) __bf16 v16bf;
typedef __attribute__((ext_vector_type(8)))  float  v8f;
typedef unsigned short ushort16_t;

__device__ __forceinline__ unsigned short f2bf(float f) {
    unsigned u = __float_as_uint(f);
    unsigned r = u + 0x7fffu + ((u >> 16) & 1u);   // round-to-nearest-even
    return (unsigned short)(r >> 16);
}

__device__ __forceinline__ void atomicMaxF(float* addr, float v) {
    if (v >= 0.0f) atomicMax((int*)addr, __float_as_int(v));
    else           atomicMin((unsigned int*)addr, __float_as_uint(v));
}

// one-time: Wt[n][k] = bf16(W[k][n])   (W: [256][512] f32 -> Wt: [512][256] bf16)
__global__ void wtrans_bf16(const float* __restrict__ W, unsigned short* __restrict__ Wt) {
    int t = blockIdx.x * blockDim.x + threadIdx.x;
    if (t >= NOUT * FDIM) return;
    int n = t >> 8, k = t & 255;
    Wt[t] = f2bf(W[(size_t)k * NOUT + n]);
}

// ---------------------------------------------------------------------------
// C[M,512] = A[M,256] @ B[256,512] (+ bias), bf16 WMMA, f32 accumulate.
// A read as f32 (converted in-register), B read pre-transposed bf16 [512][256].
// Block: 256 thr = 8 waves; each wave owns one 16x16 C tile; block tile 16x128.
// No LDS, no barriers. M is a multiple of 16 (100000, 50000).
// ---------------------------------------------------------------------------
__global__ void gemm_bf16_wmma(const float* __restrict__ A,
                               const unsigned short* __restrict__ Bt,
                               const float* __restrict__ bias,
                               float* __restrict__ C, int M) {
    const int lane = threadIdx.x & 31;
    const int wave = threadIdx.x >> 5;
    const int half = lane >> 4;
    const int l16  = lane & 15;
    const int rowBase = blockIdx.y * 16;
    const int col     = blockIdx.x * 128 + wave * 16 + l16;  // C column / Bt row

    const float*          arow = A  + (size_t)(rowBase + l16) * FDIM;
    const unsigned short* brow = Bt + (size_t)col * FDIM;

    v8f acc = {};
    #pragma unroll
    for (int k0 = 0; k0 < FDIM; k0 += 32) {
        // ---- A fragment: 16 bf16 = K[half*8 .. +7] and K[16+half*8 .. +7] ----
        const float* ap = arow + k0 + half * 8;
        float4 t0 = *(const float4*)(ap);
        float4 t1 = *(const float4*)(ap + 4);
        float4 t2 = *(const float4*)(ap + 16);
        float4 t3 = *(const float4*)(ap + 20);
        union { unsigned short s[16]; v16bf v; } fa;
        float av[16] = {t0.x, t0.y, t0.z, t0.w, t1.x, t1.y, t1.z, t1.w,
                        t2.x, t2.y, t2.z, t2.w, t3.x, t3.y, t3.z, t3.w};
        #pragma unroll
        for (int i = 0; i < 16; ++i) fa.s[i] = f2bf(av[i]);

        // ---- B fragment: 16 bf16 = K[half*8 .. +7] and K[16+half*8 .. +7] ----
        const unsigned short* bp = brow + k0 + half * 8;
        union { uint4 q[2]; v16bf v; } fb;
        fb.q[0] = *(const uint4*)(bp);
        fb.q[1] = *(const uint4*)(bp + 16);

        if (k0 + 32 < FDIM) {                 // gfx1250 global_prefetch
            __builtin_prefetch(ap + 32, 0, 1);
            __builtin_prefetch(bp + 32, 0, 1);
        }

        acc = __builtin_amdgcn_wmma_f32_16x16x32_bf16(
            false, fa.v, false, fb.v, (short)0, acc, false, false);
    }

    const float bv = bias ? bias[col] : 0.0f;
    #pragma unroll
    for (int j = 0; j < 8; ++j) {
        int r = rowBase + j + 8 * half;       // C/D layout: VGPR j -> M = j + 8*half
        C[(size_t)r * NOUT + col] = acc[j] + bv;
    }
}

// wv[f,h] = sum_d W[f, h*64+d] * a[h*64+d]    (256x8 output)
__global__ void collapse_kernel(const float* __restrict__ W,
                                const float* __restrict__ a,
                                float* __restrict__ wv) {
    int t = blockIdx.x * blockDim.x + threadIdx.x;
    if (t >= FDIM * HH) return;
    int f = t >> 3, h = t & 7;
    float s = 0.0f;
    #pragma unroll 8
    for (int d = 0; d < DD; ++d) s += W[(size_t)f * NOUT + h * DD + d] * a[h * DD + d];
    wv[f * HH + h] = s;
}

// out[n,h] = sum_f Hmat[n,f] * wv[f,h]
__global__ void eh_kernel(const float* __restrict__ Hmat,
                          const float* __restrict__ wv,
                          float* __restrict__ out, int Nn) {
    int t = blockIdx.x * blockDim.x + threadIdx.x;
    if (t >= Nn * HH) return;
    int n = t >> 3, h = t & 7;
    const float* hr = Hmat + (size_t)n * FDIM;
    float s = 0.0f;
    #pragma unroll 8
    for (int f = 0; f < FDIM; ++f) s += hr[f] * wv[f * HH + h];
    out[t] = s;
}

__global__ void fill_kernel(float* __restrict__ p, float v, int n) {
    int t = blockIdx.x * blockDim.x + threadIdx.x;
    if (t < n) p[t] = v;
}

// pass 1: leaky-relu score, store to esc, atomic max into m[dst]
__global__ void edge_max_kernel(const int* __restrict__ src, const int* __restrict__ dst,
                                const float* __restrict__ el, const float* __restrict__ er,
                                float* __restrict__ m, float* __restrict__ esc, int E) {
    int t = blockIdx.x * blockDim.x + threadIdx.x;
    if (t >= E * HH) return;
    int e = t >> 3, h = t & 7;
    float x = el[src[e] * HH + h] + er[dst[e] * HH + h];
    x = (x > 0.0f) ? x : SLOPE * x;
    esc[t] = x;
    atomicMaxF(&m[dst[e] * HH + h], x);
}

// pass 2: exp(score - max), overwrite esc, atomic-add into s[dst]
__global__ void edge_exp_kernel(const int* __restrict__ dst,
                                const float* __restrict__ m,
                                float* __restrict__ esc, float* __restrict__ ssum, int E) {
    int t = blockIdx.x * blockDim.x + threadIdx.x;
    if (t >= E * HH) return;
    int e = t >> 3, h = t & 7;
    float ex = __expf(esc[t] - m[dst[e] * HH + h]);
    esc[t] = ex;
    atomicAdd(&ssum[dst[e] * HH + h], ex);
}

// pass 3: one wave per edge; out[dst] += (esc/ssum[dst]) * fs[src]  (512-wide)
__global__ void scatter_kernel(const int* __restrict__ src, const int* __restrict__ dst,
                               const float* __restrict__ esc, const float* __restrict__ ssum,
                               const float* __restrict__ fs, float* __restrict__ out, int E) {
    int wid  = (blockIdx.x * blockDim.x + threadIdx.x) >> 5;
    int lane = threadIdx.x & 31;
    if (wid >= E) return;
    int s = src[wid], d = dst[wid];
    const float* fsrow = fs + (size_t)s * NOUT;
    float* orow = out + (size_t)d * NOUT;
    #pragma unroll 4
    for (int i = 0; i < 16; ++i) {
        int idx = i * 32 + lane;
        int h = idx >> 6;
        float a = esc[(size_t)wid * HH + h] / ssum[d * HH + h];
        atomicAdd(&orow[idx], a * fsrow[idx]);
    }
}

extern "C" void kernel_launch(void* const* d_in, const int* in_sizes, int n_in,
                              void* d_out, int out_size, void* d_ws, size_t ws_size,
                              hipStream_t stream) {
    const float* h_user    = (const float*)d_in[0];
    const float* h_item    = (const float*)d_in[1];
    const int*   src_u2i   = (const int*)d_in[2];
    const int*   dst_u2i   = (const int*)d_in[3];
    const int*   src_i2u   = (const int*)d_in[4];
    const int*   dst_i2u   = (const int*)d_in[5];
    const float* W_src_u2i = (const float*)d_in[6];
    const float* W_dst_u2i = (const float*)d_in[7];
    const float* al_u2i    = (const float*)d_in[8];
    const float* ar_u2i    = (const float*)d_in[9];
    const float* bias_u2i  = (const float*)d_in[10];
    const float* W_res_u2i = (const float*)d_in[11];
    const float* W_src_i2u = (const float*)d_in[12];
    const float* W_dst_i2u = (const float*)d_in[13];
    const float* al_i2u    = (const float*)d_in[14];
    const float* ar_i2u    = (const float*)d_in[15];
    const float* bias_i2u  = (const float*)d_in[16];
    const float* W_res_i2u = (const float*)d_in[17];

    float* out_user = (float*)d_out;                       // NU*512
    float* out_item = (float*)d_out + (size_t)NU * NOUT;   // NI*512

    // byte-carved workspace (256B aligned chunks)
    char* wsb = (char*)d_ws;
    size_t o = 0;
    auto carve = [&](size_t bytes) -> void* {
        void* p = wsb + o;
        o += (bytes + 255) & ~(size_t)255;
        return p;
    };
    float* fs_u2i  = (float*)carve((size_t)NU * NOUT * 4);
    float* fs_i2u  = (float*)carve((size_t)NI * NOUT * 4);
    float* el_u2i  = (float*)carve((size_t)NU * HH * 4);
    float* er_u2i  = (float*)carve((size_t)NI * HH * 4);
    float* el_i2u  = (float*)carve((size_t)NI * HH * 4);
    float* er_i2u  = (float*)carve((size_t)NU * HH * 4);
    float* m_u2i   = (float*)carve((size_t)NI * HH * 4);
    float* s_u2i   = (float*)carve((size_t)NI * HH * 4);
    float* m_i2u   = (float*)carve((size_t)NU * HH * 4);
    float* s_i2u   = (float*)carve((size_t)NU * HH * 4);
    float* esc_u2i = (float*)carve((size_t)E1 * HH * 4);
    float* esc_i2u = (float*)carve((size_t)E2 * HH * 4);
    float* wv0     = (float*)carve(FDIM * HH * 4);
    float* wv1     = (float*)carve(FDIM * HH * 4);
    float* wv2     = (float*)carve(FDIM * HH * 4);
    float* wv3     = (float*)carve(FDIM * HH * 4);
    unsigned short* Wt_src_u2i = (unsigned short*)carve((size_t)NOUT * FDIM * 2);
    unsigned short* Wt_src_i2u = (unsigned short*)carve((size_t)NOUT * FDIM * 2);
    unsigned short* Wt_res_u2i = (unsigned short*)carve((size_t)NOUT * FDIM * 2);
    unsigned short* Wt_res_i2u = (unsigned short*)carve((size_t)NOUT * FDIM * 2);

    // one-time weight transpose+convert (tiny)
    int wtN = NOUT * FDIM;
    wtrans_bf16<<<(wtN + 255) / 256, 256, 0, stream>>>(W_src_u2i, Wt_src_u2i);
    wtrans_bf16<<<(wtN + 255) / 256, 256, 0, stream>>>(W_src_i2u, Wt_src_i2u);
    wtrans_bf16<<<(wtN + 255) / 256, 256, 0, stream>>>(W_res_u2i, Wt_res_u2i);
    wtrans_bf16<<<(wtN + 255) / 256, 256, 0, stream>>>(W_res_i2u, Wt_res_i2u);

    // collapsed attention-vector weights: wv = einsum(W[f,h,d], a[h,d])
    collapse_kernel<<<(FDIM * HH + 127) / 128, 128, 0, stream>>>(W_src_u2i, al_u2i, wv0);
    collapse_kernel<<<(FDIM * HH + 127) / 128, 128, 0, stream>>>(W_dst_u2i, ar_u2i, wv1);
    collapse_kernel<<<(FDIM * HH + 127) / 128, 128, 0, stream>>>(W_src_i2u, al_i2u, wv2);
    collapse_kernel<<<(FDIM * HH + 127) / 128, 128, 0, stream>>>(W_dst_i2u, ar_i2u, wv3);

    // el / er per node-head
    eh_kernel<<<(NU * HH + 255) / 256, 256, 0, stream>>>(h_user, wv0, el_u2i, NU);
    eh_kernel<<<(NI * HH + 255) / 256, 256, 0, stream>>>(h_item, wv1, er_u2i, NI);
    eh_kernel<<<(NI * HH + 255) / 256, 256, 0, stream>>>(h_item, wv2, el_i2u, NI);
    eh_kernel<<<(NU * HH + 255) / 256, 256, 0, stream>>>(h_user, wv3, er_i2u, NU);

    // WMMA GEMMs: src projections into ws, residual (+bias) straight into d_out
    gemm_bf16_wmma<<<dim3(NOUT / 128, NU / 16), 256, 0, stream>>>(h_user, Wt_src_u2i, nullptr,  fs_u2i,  NU);
    gemm_bf16_wmma<<<dim3(NOUT / 128, NI / 16), 256, 0, stream>>>(h_item, Wt_src_i2u, nullptr,  fs_i2u,  NI);
    gemm_bf16_wmma<<<dim3(NOUT / 128, NI / 16), 256, 0, stream>>>(h_item, Wt_res_u2i, bias_u2i, out_item, NI);
    gemm_bf16_wmma<<<dim3(NOUT / 128, NU / 16), 256, 0, stream>>>(h_user, Wt_res_i2u, bias_i2u, out_user, NU);

    // softmax state init
    fill_kernel<<<(NI * HH + 255) / 256, 256, 0, stream>>>(m_u2i, -INFINITY, NI * HH);
    fill_kernel<<<(NI * HH + 255) / 256, 256, 0, stream>>>(s_u2i, 0.0f,      NI * HH);
    fill_kernel<<<(NU * HH + 255) / 256, 256, 0, stream>>>(m_i2u, -INFINITY, NU * HH);
    fill_kernel<<<(NU * HH + 255) / 256, 256, 0, stream>>>(s_i2u, 0.0f,      NU * HH);

    // segment softmax: max, exp-sum
    edge_max_kernel<<<(E1 * HH + 255) / 256, 256, 0, stream>>>(src_u2i, dst_u2i, el_u2i, er_u2i, m_u2i, esc_u2i, E1);
    edge_max_kernel<<<(E2 * HH + 255) / 256, 256, 0, stream>>>(src_i2u, dst_i2u, el_i2u, er_i2u, m_i2u, esc_i2u, E2);
    edge_exp_kernel<<<(E1 * HH + 255) / 256, 256, 0, stream>>>(dst_u2i, m_u2i, esc_u2i, s_u2i, E1);
    edge_exp_kernel<<<(E2 * HH + 255) / 256, 256, 0, stream>>>(dst_i2u, m_i2u, esc_i2u, s_i2u, E2);

    // weighted scatter-sum on top of residual+bias already in d_out
    scatter_kernel<<<((E1 * 32) + 255) / 256, 256, 0, stream>>>(src_u2i, dst_u2i, esc_u2i, s_u2i, fs_u2i, out_item, E1);
    scatter_kernel<<<((E2 * 32) + 255) / 256, 256, 0, stream>>>(src_i2u, dst_i2u, esc_i2u, s_i2u, fs_i2u, out_user, E2);
}
